// Mamba2LM_9234179686563
// MI455X (gfx1250) — matrix operand order repaired and checked by reading, not compiled, not run on hardware
//
#include <hip/hip_runtime.h>

// ---------------- model constants ----------------
constexpr int DM    = 256;     // d_model
constexpr int VOC   = 512;     // vocab
constexpr int DI    = 512;     // d_inner
constexpr int DSTATE= 64;      // d_state
constexpr int HD    = 32;      // headdim
constexpr int NH    = 16;      // nheads
constexpr int CHK   = 64;      // chunk
constexpr int DIP   = 1168;    // d_in_proj
constexpr int CDIM  = 640;     // conv dim
constexpr int BB    = 8;       // batch
constexpr int LL    = 4096;    // seqlen
constexpr int T     = BB*LL;   // tokens = 32768
constexpr int NBC   = BB*(LL/CHK); // 512 (b,chunk) blocks

// ---------------- types / helpers ----------------
typedef __bf16 bf16_t;
typedef __attribute__((ext_vector_type(16))) __bf16 bf16x16;
typedef __attribute__((ext_vector_type(8)))  __bf16 bf16x8;
typedef __attribute__((ext_vector_type(8)))  float  f32x8;

// native CDNA5 bf16 converts (compiler lowers fptrunc/fpext to v_cvt ops)
__device__ inline float  bf2f(bf16_t h){ return (float)h; }
__device__ inline bf16_t f2bf(float f) { return (bf16_t)f; }

__device__ inline f32x8 zero8(){
  f32x8 v = {0.f,0.f,0.f,0.f,0.f,0.f,0.f,0.f};
  return v;
}
__device__ inline f32x8 wmma_bf(bf16x16 a, bf16x16 b, f32x8 c){
  return __builtin_amdgcn_wmma_f32_16x16x32_bf16(false, a, false, b, (short)0, c, false, false);
}

// A-fragment (16x32, M x K): lane holds one row; elems 0..15 -> K = hi*16..hi*16+15
// -> two 16-byte loads.
__device__ inline bf16x16 load_a16(const bf16_t* base, int lda){
  int l = threadIdx.x & 31; int m = l & 15; int hi = l >> 4;
  const bf16_t* p = base + m*lda + hi*16;
  bf16x8 lo  = *(const bf16x8*)p;
  bf16x8 hi8 = *(const bf16x8*)(p + 8);
  bf16x16 r;
  #pragma unroll
  for (int j = 0; j < 8; ++j){ r[j] = lo[j]; r[j+8] = hi8[j]; }
  return r;
}
// B-fragment (32x16, K x N): lane = column; lanes 0-15 hold K 0..15,
// lanes 16-31 hold K 16..31 -> one 32-byte load per lane.
__device__ inline bf16x16 load_b16(const bf16_t* base, int ldb){
  int l = threadIdx.x & 31; int n = l & 15; int hi = l >> 4;
  return *(const bf16x16*)(base + n*ldb + hi*16);
}
__device__ inline bf16x16 load_b16_clamp(const bf16_t* base, int ldb, int row0, int maxrow){
  int l = threadIdx.x & 31; int n = l & 15; int hi = l >> 4;
  int r = row0 + n; if (r >= maxrow) r = maxrow - 1;
  return *(const bf16x16*)(base + (size_t)r*ldb + hi*16);
}

// ---------------- generic NT GEMM: C[m,n] = sum_k A[m,k]*B[n,k] ----------------
// block = 256 thr (8 waves), tile 128(M) x 64(N); wave -> 32x32 (2x2 wmma tiles)
// EPI 0: f32 store.  EPI 1: bf16 store of (acc + resid[m*ldc+n]).
template<int EPI>
__global__ __launch_bounds__(256)
void gemm_nt(const bf16_t* __restrict__ A, const bf16_t* __restrict__ B,
             float* __restrict__ Cf, bf16_t* __restrict__ Cb,
             const float* __restrict__ resid,
             int M, int N, int K, int ldc)
{
  const int wave = threadIdx.x >> 5;
  const int wm = wave & 3, wn = wave >> 2;
  const int m0 = blockIdx.y*128 + wm*32;
  const int n0 = blockIdx.x*64  + wn*32;
  f32x8 acc00 = zero8(), acc01 = zero8(), acc10 = zero8(), acc11 = zero8();
  for (int k0 = 0; k0 < K; k0 += 32){
    bf16x16 a0 = load_a16(A + (size_t)m0*K + k0, K);
    bf16x16 a1 = load_a16(A + (size_t)(m0+16)*K + k0, K);
    bf16x16 b0 = load_b16_clamp(B + k0, K, n0,      N);
    bf16x16 b1 = load_b16_clamp(B + k0, K, n0 + 16, N);
    acc00 = wmma_bf(a0, b0, acc00);
    acc01 = wmma_bf(a0, b1, acc01);
    acc10 = wmma_bf(a1, b0, acc10);
    acc11 = wmma_bf(a1, b1, acc11);
    if (k0 + 32 < K){
      __builtin_prefetch(A + (size_t)m0*K + k0 + 32, 0, 3);   // global_prefetch_b8
      __builtin_prefetch(B + (size_t)n0*K + k0 + 32, 0, 3);
    }
  }
  const int l = threadIdx.x & 31, cn = l & 15, hi = l >> 4;
  f32x8 accs[2][2] = {{acc00, acc01},{acc10, acc11}};
  #pragma unroll
  for (int i = 0; i < 2; ++i)
    #pragma unroll
    for (int j = 0; j < 2; ++j){
      int n = n0 + j*16 + cn;
      if (n >= N) continue;
      #pragma unroll
      for (int r = 0; r < 8; ++r){
        int m = m0 + i*16 + r + hi*8;
        size_t idx = (size_t)m*ldc + n;
        float v = accs[i][j][r];
        if (EPI == 0) Cf[idx] = v;
        else          Cb[idx] = f2bf(v + resid[idx]);
      }
    }
}

// ---------------- small elementwise kernels ----------------
__global__ __launch_bounds__(256)
void k_f2bf(const float* __restrict__ s, bf16_t* __restrict__ d, int n){
  int i = blockIdx.x*256 + threadIdx.x;
  if (i < n) d[i] = f2bf(s[i]);
}

// embedding gather + rmsnorm(norm_w, 1e-6): x_emb f32, u bf16
__global__ __launch_bounds__(256)
void k_embed_rms(const int* __restrict__ ids, const float* __restrict__ embed,
                 const float* __restrict__ nw,
                 float* __restrict__ x_emb, bf16_t* __restrict__ u)
{
  int tok = blockIdx.x, ch = threadIdx.x;
  float e = embed[(size_t)ids[tok]*DM + ch];
  __shared__ float red[256];
  red[ch] = e*e; __syncthreads();
  for (int s = 128; s > 0; s >>= 1){ if (ch < s) red[ch] += red[ch+s]; __syncthreads(); }
  float sc = rsqrtf(red[0]/(float)DM + 1e-6f);
  x_emb[(size_t)tok*DM + ch] = e;
  u[(size_t)tok*DM + ch]     = f2bf(e*sc*nw[ch]);
}

// dt_sp = softplus(dt + dt_bias)
__global__ __launch_bounds__(256)
void k_dt(const float* __restrict__ zx, const float* __restrict__ dtb,
          float* __restrict__ dtsp)
{
  int i = blockIdx.x*256 + threadIdx.x;
  if (i >= T*NH) return;
  int tok = i >> 4, h = i & 15;
  float x = zx[(size_t)tok*DIP + DI + CDIM + h] + dtb[h];
  dtsp[i] = (x > 20.f) ? x : log1pf(expf(x));
}

// per (b,chunk,head): inclusive cumsum over q of dt_sp * (-exp(A_log))
__global__ __launch_bounds__(64)
void k_acs(const float* __restrict__ dtsp, const float* __restrict__ Alog,
           float* __restrict__ acs)
{
  int bc = blockIdx.x >> 4, h = blockIdx.x & 15, q = threadIdx.x;
  float Ah = -expf(Alog[h]);
  __shared__ float s[64];
  s[q] = dtsp[(size_t)(bc*CHK + q)*NH + h] * Ah; __syncthreads();
  for (int off = 1; off < 64; off <<= 1){
    float t = (q >= off) ? s[q-off] : 0.f;
    __syncthreads();
    s[q] += t;
    __syncthreads();
  }
  acs[((size_t)bc*NH + h)*CHK + q] = s[q];
}

// causal depthwise conv(4) + bias + silu; split into x / B / C; Xd = x*dt
__global__ __launch_bounds__(256)
void k_conv(const float* __restrict__ zx, const float* __restrict__ cw,
            const float* __restrict__ cb, const float* __restrict__ dtsp,
            float* __restrict__ xconv, bf16_t* __restrict__ xd,
            bf16_t* __restrict__ bm, bf16_t* __restrict__ cm)
{
  int i = blockIdx.x*256 + threadIdx.x;
  if (i >= T*CDIM) return;
  int ch = i % CDIM, tok = i / CDIM;
  int l = tok & (LL-1);
  float acc = cb[ch];
  #pragma unroll
  for (int j = 0; j < 4; ++j){
    int ll = l - 3 + j;
    if (ll >= 0) acc += cw[ch*4 + j] * zx[(size_t)(tok - 3 + j)*DIP + DI + ch];
  }
  float s = acc / (1.f + expf(-acc));      // silu
  if (ch < DI){
    int h = ch >> 5;
    xconv[(size_t)tok*DI + ch] = s;
    xd[(size_t)tok*DI + ch]    = f2bf(s * dtsp[(size_t)tok*NH + h]);
  } else if (ch < DI + DSTATE){
    bm[(size_t)tok*DSTATE + (ch - DI)] = f2bf(s);
  } else {
    cm[(size_t)tok*DSTATE + (ch - DI - DSTATE)] = f2bf(s);
  }
}

// ---------------- SSD chunk kernel: G, Y_diag(+skip), per-chunk states ----------------
// one block per (b,chunk); 8 waves; wave w handles heads w and w+8
__global__ __launch_bounds__(256)
void k_chunk(const bf16_t* __restrict__ bm, const bf16_t* __restrict__ cm,
             const float* __restrict__ acs, const bf16_t* __restrict__ xd,
             const float* __restrict__ xconv, const float* __restrict__ Dsk,
             float* __restrict__ Y, float* __restrict__ states)
{
  __shared__ __attribute__((aligned(32))) bf16_t sB[64*64];     // Bm[k][n]
  __shared__ __attribute__((aligned(32))) bf16_t sC[64*64];     // Cm[q][n]
  __shared__ __attribute__((aligned(32))) bf16_t sG[64*64];     // G[q][k]
  __shared__ float  sAcs[16*64];
  __shared__ __attribute__((aligned(32))) bf16_t sXdT[8][32*64]; // per-wave Xd^T [p][k]
  __shared__ float  sDec[8][64];

  const int bc = blockIdx.x;
  const size_t tok0 = (size_t)bc*64;          // b*4096 + c*64 == bc*64
  const int wave = threadIdx.x >> 5, lane = threadIdx.x & 31;
  const int cn = lane & 15, hi = lane >> 4;

  for (int i = threadIdx.x; i < 64*64; i += 256){
    sB[i] = bm[(tok0 + (i >> 6))*DSTATE + (i & 63)];
    sC[i] = cm[(tok0 + (i >> 6))*DSTATE + (i & 63)];
  }
  for (int i = threadIdx.x; i < 16*64; i += 256)
    sAcs[i] = acs[(size_t)bc*16*64 + i];
  __syncthreads();

  // G = C * B^T  (q x k), contraction over n=64
  for (int tt = 0; tt < 2; ++tt){
    int t  = wave*2 + tt;
    int qt = t >> 2, kt = t & 3;
    f32x8 acc = zero8();
    #pragma unroll
    for (int kf = 0; kf < 2; ++kf){
      bf16x16 a = load_a16(sC + qt*16*64 + kf*32, 64);
      bf16x16 b = load_b16(sB + kt*16*64 + kf*32, 64);
      acc = wmma_bf(a, b, acc);
    }
    int kcol = kt*16 + cn;
    #pragma unroll
    for (int r = 0; r < 8; ++r)
      sG[(qt*16 + r + hi*8)*64 + kcol] = f2bf(acc[r]);
  }
  __syncthreads();

  for (int hp = 0; hp < 2; ++hp){
    int h = wave + hp*8;
    const float* Ac = sAcs + h*64;
    for (int i = lane; i < 64; i += 32)
      sDec[wave][i] = expf(Ac[63] - Ac[i]);
    for (int i = lane; i < 2048; i += 32){
      int k = i >> 5, p = i & 31;
      sXdT[wave][p*64 + k] = xd[(tok0 + k)*DI + h*32 + p];
    }

    // Y_diag = (exp(segsum) o G) @ Xd   (64x32, K=64)
    for (int mt = 0; mt < 4; ++mt){
      int mq = mt*16 + cn;                 // row this lane owns
      bf16x16 af[2];
      #pragma unroll
      for (int kt = 0; kt < 2; ++kt){
        bf16x16 a;
        #pragma unroll
        for (int j = 0; j < 16; ++j){
          int kk = kt*32 + hi*16 + j;
          float v = 0.f;
          if (kk <= mq) v = bf2f(sG[mq*64 + kk]) * expf(Ac[mq] - Ac[kk]);
          a[j] = f2bf(v);
        }
        af[kt] = a;
      }
      #pragma unroll
      for (int nt = 0; nt < 2; ++nt){
        f32x8 acc = zero8();
        acc = wmma_bf(af[0], load_b16(sXdT[wave] + nt*16*64 + 0,  64), acc);
        acc = wmma_bf(af[1], load_b16(sXdT[wave] + nt*16*64 + 32, 64), acc);
        int p = nt*16 + cn;
        #pragma unroll
        for (int r = 0; r < 8; ++r){
          int q = mt*16 + r + hi*8;
          size_t idx = (tok0 + q)*DI + h*32 + p;
          Y[idx] = acc[r] + Dsk[h]*xconv[idx];
        }
      }
    }

    // states[n][p] = sum_k Bm[k][n]*exp(Acs[63]-Acs[k])*Xd[k][p]
    for (int ntl = 0; ntl < 4; ++ntl){
      int nn = ntl*16 + cn;
      bf16x16 af[2];
      #pragma unroll
      for (int kt = 0; kt < 2; ++kt){
        bf16x16 a;
        #pragma unroll
        for (int j = 0; j < 16; ++j){
          int kk = kt*32 + hi*16 + j;
          a[j] = f2bf(bf2f(sB[kk*64 + nn]) * sDec[wave][kk]);
        }
        af[kt] = a;
      }
      #pragma unroll
      for (int pt = 0; pt < 2; ++pt){
        f32x8 acc = zero8();
        acc = wmma_bf(af[0], load_b16(sXdT[wave] + pt*16*64 + 0,  64), acc);
        acc = wmma_bf(af[1], load_b16(sXdT[wave] + pt*16*64 + 32, 64), acc);
        int p = pt*16 + cn;
        #pragma unroll
        for (int r = 0; r < 8; ++r){
          int n = ntl*16 + r + hi*8;
          states[(((size_t)bc*16 + h)*64 + n)*32 + p] = acc[r];
        }
      }
    }
  }
}

// sequential inter-chunk scan, in place: states[c] <- prev-state before chunk c
__global__ __launch_bounds__(256)
void k_scan(const float* __restrict__ acs, float* __restrict__ states)
{
  int b = blockIdx.x >> 4, h = blockIdx.x & 15;
  int e0 = threadIdx.x * 8;
  float run[8];
  #pragma unroll
  for (int j = 0; j < 8; ++j) run[j] = 0.f;
  for (int c = 0; c < 64; ++c){
    size_t bch = (size_t)(b*64 + c)*16 + h;
    float dec = expf(acs[bch*64 + 63]);
    float* p = states + bch*2048 + e0;
    #pragma unroll
    for (int j = 0; j < 8; ++j){
      float t = p[j];
      p[j] = run[j];
      run[j] = dec*run[j] + t;
    }
  }
}

// Y += exp(Acs[q]) * C @ prev^T
__global__ __launch_bounds__(256)
void k_yoff(const bf16_t* __restrict__ cm, const float* __restrict__ acs,
            const float* __restrict__ prev, float* __restrict__ Y)
{
  __shared__ __attribute__((aligned(32))) bf16_t sC[64*64];
  __shared__ float sAcs[16*64];
  __shared__ __attribute__((aligned(32))) bf16_t sPT[8][32*64];  // prev^T [p][n]
  const int bc = blockIdx.x;
  const size_t tok0 = (size_t)bc*64;
  const int wave = threadIdx.x >> 5, lane = threadIdx.x & 31;
  const int cn = lane & 15, hi = lane >> 4;

  for (int i = threadIdx.x; i < 64*64; i += 256)
    sC[i] = cm[(tok0 + (i >> 6))*DSTATE + (i & 63)];
  for (int i = threadIdx.x; i < 16*64; i += 256)
    sAcs[i] = acs[(size_t)bc*16*64 + i];
  __syncthreads();

  for (int hp = 0; hp < 2; ++hp){
    int h = wave + hp*8;
    const float* pv = prev + ((size_t)bc*16 + h)*2048;
    for (int i = lane; i < 2048; i += 32){
      int n = i >> 5, p = i & 31;
      sPT[wave][p*64 + n] = f2bf(pv[i]);
    }
    const float* Ac = sAcs + h*64;
    for (int mt = 0; mt < 4; ++mt){
      bf16x16 a0 = load_a16(sC + mt*16*64 + 0,  64);
      bf16x16 a1 = load_a16(sC + mt*16*64 + 32, 64);
      #pragma unroll
      for (int pt = 0; pt < 2; ++pt){
        f32x8 acc = zero8();
        acc = wmma_bf(a0, load_b16(sPT[wave] + pt*16*64 + 0,  64), acc);
        acc = wmma_bf(a1, load_b16(sPT[wave] + pt*16*64 + 32, 64), acc);
        int p = pt*16 + cn;
        #pragma unroll
        for (int r = 0; r < 8; ++r){
          int q = mt*16 + r + hi*8;
          size_t idx = (tok0 + q)*DI + h*32 + p;
          Y[idx] += expf(Ac[q]) * acc[r];
        }
      }
    }
  }
}

// gate with silu(z), rmsnorm(gate_norm_w, 1e-5), to bf16; 1 token per wave
__global__ __launch_bounds__(256)
void k_gate(const float* __restrict__ Y, const float* __restrict__ zx,
            const float* __restrict__ gw, bf16_t* __restrict__ yg)
{
  int wave = threadIdx.x >> 5, lane = threadIdx.x & 31;
  size_t tok = (size_t)blockIdx.x*8 + wave;
  float v[16]; float ss = 0.f;
  #pragma unroll
  for (int i = 0; i < 16; ++i){
    int ch = lane + i*32;
    float z = zx[tok*DIP + ch];
    float y = Y[tok*DI + ch] * (z / (1.f + expf(-z)));
    v[i] = y; ss += y*y;
  }
  #pragma unroll
  for (int o = 16; o > 0; o >>= 1) ss += __shfl_xor(ss, o, 32);
  float sc = rsqrtf(ss/(float)DI + 1e-5f);
  #pragma unroll
  for (int i = 0; i < 16; ++i){
    int ch = lane + i*32;
    yg[tok*DI + ch] = f2bf(v[i]*sc*gw[ch]);
  }
}

// ---------------- host launch ----------------
extern "C" void kernel_launch(void* const* d_in, const int* in_sizes, int n_in,
                              void* d_out, int out_size, void* d_ws, size_t ws_size,
                              hipStream_t stream)
{
  const int*   ids   = (const int*)  d_in[0];
  const float* embed = (const float*)d_in[1];
  const float* inpw  = (const float*)d_in[2];
  const float* convw = (const float*)d_in[3];
  const float* convb = (const float*)d_in[4];
  const float* Alog  = (const float*)d_in[5];
  const float* Dsk   = (const float*)d_in[6];
  const float* dtb   = (const float*)d_in[7];
  const float* gnw   = (const float*)d_in[8];
  const float* outpw = (const float*)d_in[9];
  const float* nw    = (const float*)d_in[10];

  char* ws = (char*)d_ws;
  size_t off = 0;
  auto alloc = [&](size_t bytes)->char*{
    char* p = ws + off; off += (bytes + 255) & ~(size_t)255; return p;
  };
  float*  x_emb  = (float*) alloc((size_t)T*DM*4);          //  33.5 MB
  bf16_t* u_bf   = (bf16_t*)alloc((size_t)T*DM*2);          //  16.8 MB
  bf16_t* w_inp  = (bf16_t*)alloc((size_t)DIP*DM*2);
  bf16_t* w_outp = (bf16_t*)alloc((size_t)DM*DI*2);
  bf16_t* w_emb  = (bf16_t*)alloc((size_t)VOC*DM*2);
  float*  zx     = (float*) alloc((size_t)T*DIP*4);         // 153.1 MB
  float*  dtsp   = (float*) alloc((size_t)T*NH*4);
  float*  acs    = (float*) alloc((size_t)NBC*NH*CHK*4);
  float*  xconv  = (float*) alloc((size_t)T*DI*4);          //  67.1 MB
  bf16_t* xd     = (bf16_t*)alloc((size_t)T*DI*2);          //  33.5 MB
  bf16_t* bm     = (bf16_t*)alloc((size_t)T*DSTATE*2);
  bf16_t* cmb    = (bf16_t*)alloc((size_t)T*DSTATE*2);
  float*  Y      = (float*) alloc((size_t)T*DI*4);          //  67.1 MB
  float*  states = (float*) alloc((size_t)NBC*NH*DSTATE*HD*4); // 67.1 MB
  bf16_t* yg     = (bf16_t*)alloc((size_t)T*DI*2);          //  33.5 MB
  bf16_t* hid    = (bf16_t*)alloc((size_t)T*DM*2);          //  16.8 MB
  (void)in_sizes; (void)n_in; (void)out_size; (void)ws_size;

  // weight conversions
  k_f2bf<<<(DIP*DM+255)/256, 256, 0, stream>>>(inpw,  w_inp,  DIP*DM);
  k_f2bf<<<(DM*DI +255)/256, 256, 0, stream>>>(outpw, w_outp, DM*DI);
  k_f2bf<<<(VOC*DM+255)/256, 256, 0, stream>>>(embed, w_emb,  VOC*DM);

  // embedding + rmsnorm
  k_embed_rms<<<T, 256, 0, stream>>>(ids, embed, nw, x_emb, u_bf);

  // in_proj: (T,256)x(1168,256)^T -> zx (T,1168) f32
  gemm_nt<0><<<dim3((DIP+63)/64, T/128), 256, 0, stream>>>(
      u_bf, w_inp, zx, nullptr, nullptr, T, DIP, DM, DIP);

  // dt softplus, conv+silu+split, cumsum A
  k_dt  <<<(T*NH+255)/256, 256, 0, stream>>>(zx, dtb, dtsp);
  k_conv<<<(T*CDIM)/256,   256, 0, stream>>>(zx, convw, convb, dtsp, xconv, xd, bm, cmb);
  k_acs <<<NBC*NH, 64, 0, stream>>>(dtsp, Alog, acs);

  // SSD: diagonal part + per-chunk states, inter-chunk scan, off-diagonal part
  k_chunk<<<NBC, 256, 0, stream>>>(bm, cmb, acs, xd, xconv, Dsk, Y, states);
  k_scan <<<BB*NH, 256, 0, stream>>>(acs, states);
  k_yoff <<<NBC, 256, 0, stream>>>(cmb, acs, states, Y);

  // gate + rmsnorm -> bf16
  k_gate<<<T/8, 256, 0, stream>>>(Y, zx, gnw, yg);

  // out_proj with fused residual: hid = bf16(x_emb + yg @ out_proj_w^T)
  gemm_nt<1><<<dim3(DM/64, T/128), 256, 0, stream>>>(
      yg, w_outp, nullptr, hid, x_emb, T, DM, DI, DM);

  // logits = hid @ embed^T -> d_out f32 (T,512)
  gemm_nt<0><<<dim3(VOC/64, T/128), 256, 0, stream>>>(
      hid, w_emb, (float*)d_out, nullptr, nullptr, T, VOC, DM, VOC);
}
